// GCNII_39058432590072
// MI455X (gfx1250) — matrix operand order repaired and checked by reading, not compiled, not run on hardware
//
#include <hip/hip_runtime.h>
#include <cstddef>
#include <cstdint>

#define N_NODES 100000
#define N_EDGES 1600000
#define DIM 128
#define NCLS 64
#define ALPHA 0.1f

typedef __attribute__((ext_vector_type(16))) __bf16 v16bf;
typedef __attribute__((ext_vector_type(8)))  float  v8f;
typedef __attribute__((ext_vector_type(4)))  int    v4i;

#if __has_builtin(__builtin_amdgcn_global_load_async_to_lds_b128) && \
    __has_builtin(__builtin_amdgcn_s_wait_asynccnt)
#define USE_ASYNC_LDS 1
#endif

static __device__ __forceinline__ unsigned short bf16bits(float f) {
    return __builtin_bit_cast(unsigned short, (__bf16)f);
}

static __device__ __forceinline__ v8f wmma_bf16(v16bf a, v16bf b, v8f c) {
    return __builtin_amdgcn_wmma_f32_16x16x32_bf16(false, a, false, b, (short)0, c, false, false);
}

// Copy S bf16 values of pre-transposed weight from global to LDS (16B per lane).
static __device__ __forceinline__ void stage_weights(const unsigned short* __restrict__ Wbf,
                                                     unsigned short* __restrict__ sW,
                                                     int S, int nthreads) {
#ifdef USE_ASYNC_LDS
    for (int off = threadIdx.x * 8; off < S; off += nthreads * 8) {
        __builtin_amdgcn_global_load_async_to_lds_b128(
            (v4i*)(Wbf + off), (v4i*)(sW + off), 0, 0);
    }
    __builtin_amdgcn_s_wait_asynccnt(0);
#else
    for (int i = threadIdx.x; i < S / 8; i += nthreads)
        ((uint4*)sW)[i] = ((const uint4*)Wbf)[i];
#endif
    __syncthreads();
}

// A-fragment: 16x32 tile of row-major fp32 matrix (ld=128) -> bf16 WMMA-A layout.
static __device__ __forceinline__ v16bf load_a(const float* __restrict__ M,
                                               int row0, int kc, int lane) {
    int m = lane & 15, hh = lane >> 4;
    const float* rp = M + (size_t)(row0 + m) * DIM + kc + 8 * hh;
    v16bf a;
#pragma unroll
    for (int i = 0; i < 4; ++i) {
        float2 v = *(const float2*)(rp + 2 * i);
        a[2 * i]     = (__bf16)v.x;
        a[2 * i + 1] = (__bf16)v.y;
        float2 w = *(const float2*)(rp + 16 + 2 * i);
        a[8 + 2 * i] = (__bf16)w.x;
        a[9 + 2 * i] = (__bf16)w.y;
    }
    return a;
}

// A-fragment of support = (1-ALPHA)*hi + ALPHA*h0, mixed on the fly.
static __device__ __forceinline__ v16bf load_a_support(const float* __restrict__ hi,
                                                       const float* __restrict__ h0,
                                                       int row0, int kc, int lane) {
    int m = lane & 15, hh = lane >> 4;
    size_t base = (size_t)(row0 + m) * DIM + kc + 8 * hh;
    const float* pi = hi + base;
    const float* p0 = h0 + base;
    v16bf a;
#pragma unroll
    for (int i = 0; i < 4; ++i) {
        float2 vi = *(const float2*)(pi + 2 * i);
        float2 v0 = *(const float2*)(p0 + 2 * i);
        a[2 * i]     = (__bf16)((1.f - ALPHA) * vi.x + ALPHA * v0.x);
        a[2 * i + 1] = (__bf16)((1.f - ALPHA) * vi.y + ALPHA * v0.y);
        float2 wi = *(const float2*)(pi + 16 + 2 * i);
        float2 w0 = *(const float2*)(p0 + 16 + 2 * i);
        a[8 + 2 * i] = (__bf16)((1.f - ALPHA) * wi.x + ALPHA * w0.x);
        a[9 + 2 * i] = (__bf16)((1.f - ALPHA) * wi.y + ALPHA * w0.y);
    }
    return a;
}

// B-fragment from LDS: W stored transposed as bf16 bits, Wt[col*128 + k].
static __device__ __forceinline__ v16bf load_b(const unsigned short* __restrict__ Wt,
                                               int col0, int kc, int lane) {
    int n = lane & 15, hh = lane >> 4;
    const unsigned short* bp = Wt + (size_t)(col0 + n) * DIM + kc + 8 * hh;
    v16bf b;
#pragma unroll
    for (int v = 0; v < 8; ++v) {
        b[2 * v]     = __builtin_bit_cast(__bf16, bp[2 * v + (v >= 4 ? 8 : 0)]);
        b[2 * v + 1] = __builtin_bit_cast(__bf16, bp[2 * v + 1 + (v >= 4 ? 8 : 0)]);
    }
    return b;
}

// ---------------- weight prep: fp32 row-major [128 x C] -> bf16 transposed [C x 128] --
__global__ void __launch_bounds__(256) prep_w_kernel(const float* __restrict__ src,
                                                     unsigned short* __restrict__ dst,
                                                     int C) {
    int idx = blockIdx.x * 256 + threadIdx.x;
    if (idx >= DIM * C) return;
    int k = idx / C, c = idx % C;
    dst[c * DIM + k] = bf16bits(src[idx]);
}

// ---------------- h0 = relu(x @ w_in + b_in) ----------------
// 8 waves/block, each wave: 16 rows x all 128 cols (8 WMMA tiles, A reused).
__global__ void __launch_bounds__(256) gemm_in_kernel(const float* __restrict__ x,
                                                      const unsigned short* __restrict__ Wbf,
                                                      const float* __restrict__ b_in,
                                                      float* __restrict__ h0) {
    __shared__ unsigned short sW[DIM * DIM];   // 32 KB
    stage_weights(Wbf, sW, DIM * DIM, 256);

    int wv = threadIdx.x >> 5, lane = threadIdx.x & 31;
    int row0 = blockIdx.x * 128 + wv * 16;
    if (row0 >= N_NODES) return;               // wave-uniform tail guard

    v8f acc[8];
#pragma unroll
    for (int t = 0; t < 8; ++t) acc[t] = (v8f){};
#pragma unroll
    for (int kc = 0; kc < DIM; kc += 32) {
        v16bf a = load_a(x, row0, kc, lane);
#pragma unroll
        for (int t = 0; t < 8; ++t) {
            v16bf b = load_b(sW, t * 16, kc, lane);
            acc[t] = wmma_bf16(a, b, acc[t]);
        }
    }
    int n = lane & 15, hh = lane >> 4;
#pragma unroll
    for (int t = 0; t < 8; ++t) {
        float bias = b_in[t * 16 + n];
#pragma unroll
        for (int r = 0; r < 8; ++r) {
            int row = row0 + r + 8 * hh;
            h0[(size_t)row * DIM + t * 16 + n] = fmaxf(acc[t][r] + bias, 0.f);
        }
    }
}

// ---------------- zero-fill hi ----------------
__global__ void __launch_bounds__(256) fill_zero_kernel(float4* __restrict__ p) {
    size_t i = (size_t)blockIdx.x * 256 + threadIdx.x;
    p[i] = make_float4(0.f, 0.f, 0.f, 0.f);
}

// ---------------- hi[dst] += w * h[src]  (SpMM scatter, L2-resident atomics) ------
__global__ void __launch_bounds__(256) spmm_kernel(const float* __restrict__ h,
                                                   const int* __restrict__ src,
                                                   const int* __restrict__ dst,
                                                   const float* __restrict__ ew,
                                                   float* __restrict__ hi) {
    size_t idx = (size_t)blockIdx.x * 256 + threadIdx.x;
    int e = (int)(idx >> 5);
    if (e >= N_EDGES) return;
    int q = (int)(idx & 31);
    int s = src[e], d = dst[e];
    float w = ew[e];
    float4 v = *(const float4*)(h + (size_t)s * DIM + q * 4);
    float* o = hi + (size_t)d * DIM + q * 4;
    atomicAdd(o + 0, w * v.x);
    atomicAdd(o + 1, w * v.y);
    atomicAdd(o + 2, w * v.z);
    atomicAdd(o + 3, w * v.w);
}

// -------- h_next = relu(theta*(support@W) + (1-theta)*support + h_cur) ------------
__global__ void __launch_bounds__(256) gemm_layer_kernel(const float* __restrict__ hi,
                                                         const float* __restrict__ h0,
                                                         const float* __restrict__ hcur,
                                                         const unsigned short* __restrict__ Wbf,
                                                         float theta,
                                                         float* __restrict__ hout) {
    __shared__ unsigned short sW[DIM * DIM];
    stage_weights(Wbf, sW, DIM * DIM, 256);

    int wv = threadIdx.x >> 5, lane = threadIdx.x & 31;
    int row0 = blockIdx.x * 128 + wv * 16;
    if (row0 >= N_NODES) return;

    v8f acc[8];
#pragma unroll
    for (int t = 0; t < 8; ++t) acc[t] = (v8f){};
#pragma unroll
    for (int kc = 0; kc < DIM; kc += 32) {
        v16bf a = load_a_support(hi, h0, row0, kc, lane);
#pragma unroll
        for (int t = 0; t < 8; ++t) {
            v16bf b = load_b(sW, t * 16, kc, lane);
            acc[t] = wmma_bf16(a, b, acc[t]);
        }
    }
    int n = lane & 15, hh = lane >> 4;
#pragma unroll
    for (int t = 0; t < 8; ++t) {
#pragma unroll
        for (int r = 0; r < 8; ++r) {
            int row = row0 + r + 8 * hh;
            size_t off = (size_t)row * DIM + t * 16 + n;
            float sup = (1.f - ALPHA) * hi[off] + ALPHA * h0[off];
            float o = theta * acc[t][r] + (1.f - theta) * sup + hcur[off];
            hout[off] = fmaxf(o, 0.f);
        }
    }
}

// ---------------- logits = h @ w_out + b_out ----------------
__global__ void __launch_bounds__(256) gemm_out_kernel(const float* __restrict__ h,
                                                       const unsigned short* __restrict__ Wbf,
                                                       const float* __restrict__ b_out,
                                                       float* __restrict__ logits) {
    __shared__ unsigned short sW[NCLS * DIM];  // 16 KB
    stage_weights(Wbf, sW, NCLS * DIM, 256);

    int wv = threadIdx.x >> 5, lane = threadIdx.x & 31;
    int row0 = blockIdx.x * 128 + wv * 16;
    if (row0 >= N_NODES) return;

    v8f acc[4];
#pragma unroll
    for (int t = 0; t < 4; ++t) acc[t] = (v8f){};
#pragma unroll
    for (int kc = 0; kc < DIM; kc += 32) {
        v16bf a = load_a(h, row0, kc, lane);
#pragma unroll
        for (int t = 0; t < 4; ++t) {
            v16bf b = load_b(sW, t * 16, kc, lane);
            acc[t] = wmma_bf16(a, b, acc[t]);
        }
    }
    int n = lane & 15, hh = lane >> 4;
#pragma unroll
    for (int t = 0; t < 4; ++t) {
        float bias = b_out[t * 16 + n];
#pragma unroll
        for (int r = 0; r < 8; ++r) {
            int row = row0 + r + 8 * hh;
            logits[(size_t)row * NCLS + t * 16 + n] = acc[t][r] + bias;
        }
    }
}

// ---------------- in-place row-wise log_softmax over 64 classes ----------------
__global__ void __launch_bounds__(256) logsoftmax_kernel(float* __restrict__ logits) {
    int wv = threadIdx.x >> 5, lane = threadIdx.x & 31;
    int row = blockIdx.x * 8 + wv;
    if (row >= N_NODES) return;
    float* p = logits + (size_t)row * NCLS;
    float v0 = p[lane], v1 = p[lane + 32];
    float m = fmaxf(v0, v1);
#pragma unroll
    for (int o = 16; o; o >>= 1) m = fmaxf(m, __shfl_xor(m, o, 32));
    float s = __expf(v0 - m) + __expf(v1 - m);
#pragma unroll
    for (int o = 16; o; o >>= 1) s += __shfl_xor(s, o, 32);
    float lse = m + __logf(s);
    p[lane]      = v0 - lse;
    p[lane + 32] = v1 - lse;
}

extern "C" void kernel_launch(void* const* d_in, const int* in_sizes, int n_in,
                              void* d_out, int out_size, void* d_ws, size_t ws_size,
                              hipStream_t stream) {
    const float* x      = (const float*)d_in[0];
    const int*   esrc   = (const int*)d_in[1];
    const int*   edst   = (const int*)d_in[2];
    const float* ew     = (const float*)d_in[3];
    const float* w_in   = (const float*)d_in[4];
    const float* b_in   = (const float*)d_in[5];
    const float* gcn_w  = (const float*)d_in[6];
    const float* w_out  = (const float*)d_in[7];
    const float* b_out  = (const float*)d_in[8];
    float* out = (float*)d_out;

    const size_t NH = (size_t)N_NODES * DIM;
    float* h0 = (float*)d_ws;
    float* hA = h0 + NH;
    float* hB = hA + NH;
    float* hi = hB + NH;
    // pre-converted transposed bf16 weights after the 4 big buffers
    unsigned short* wIn  = (unsigned short*)(hi + NH);
    unsigned short* wGcn = wIn + DIM * DIM;               // 4 matrices
    unsigned short* wOut = wGcn + 4 * DIM * DIM;

    // theta_l = log(0.5/(l+2) + 1)
    const float theta[4] = {0.22314355f, 0.15415068f, 0.11778304f, 0.09531018f};

    const int rowBlocks  = (N_NODES + 127) / 128;              // 782
    const int fillBlocks = (int)(NH / 4 / 256);                // 12500
    const int spmmBlocks = (int)(((size_t)N_EDGES * 32) / 256);// 200000

    // weight prep (tiny)
    prep_w_kernel<<<(DIM * DIM + 255) / 256, 256, 0, stream>>>(w_in, wIn, DIM);
    for (int l = 0; l < 4; ++l)
        prep_w_kernel<<<(DIM * DIM + 255) / 256, 256, 0, stream>>>(
            gcn_w + (size_t)l * DIM * DIM, wGcn + (size_t)l * DIM * DIM, DIM);
    prep_w_kernel<<<(DIM * NCLS + 255) / 256, 256, 0, stream>>>(w_out, wOut, NCLS);

    gemm_in_kernel<<<rowBlocks, 256, 0, stream>>>(x, wIn, b_in, h0);

    const float* hc = h0;
    float* buf[2] = {hA, hB};
    for (int l = 0; l < 4; ++l) {
        float* hn = buf[l & 1];
        fill_zero_kernel<<<fillBlocks, 256, 0, stream>>>((float4*)hi);
        spmm_kernel<<<spmmBlocks, 256, 0, stream>>>(hc, esrc, edst, ew, hi);
        gemm_layer_kernel<<<rowBlocks, 256, 0, stream>>>(hi, h0, hc,
                                                         wGcn + (size_t)l * DIM * DIM,
                                                         theta[l], hn);
        hc = hn;
    }

    gemm_out_kernel<<<rowBlocks, 256, 0, stream>>>(hc, wOut, b_out, out);
    logsoftmax_kernel<<<N_NODES / 8, 256, 0, stream>>>(out);
}
